// AttnBlock_20109036880013
// MI455X (gfx1250) — matrix operand order repaired and checked by reading, not compiled, not run on hardware
//
#include <hip/hip_runtime.h>
#include <hip/hip_bf16.h>
#include <math.h>

// Problem constants (from reference setup_inputs)
#define BATCH 4
#define CCH   512
#define HWSZ  4096
#define TCH   1536          // 3*C
#define GROUPS 32
#define RSQRT_C 0.044194173824159216f   // 512^-0.5

typedef __attribute__((ext_vector_type(16))) __bf16 v16bf;
typedef __attribute__((ext_vector_type(8)))  __bf16 v8bf;
typedef __attribute__((ext_vector_type(8)))  float  v8f;
typedef __attribute__((ext_vector_type(4)))  int    v4i;

// ---- CDNA5 async global->LDS path (guarded; fallback = sync copy) ---------
#if defined(__has_builtin)
#  if __has_builtin(__builtin_amdgcn_global_load_async_to_lds_b128) && \
      __has_builtin(__builtin_amdgcn_s_wait_asynccnt)
#    define HAVE_ASYNC_LDS 1
#  endif
#endif
#ifndef HAVE_ASYNC_LDS
#  define HAVE_ASYNC_LDS 0
#endif

#if HAVE_ASYNC_LDS
// Probed signature: (AS1 int4* src, AS3 int4* lds_dst, imm offset, imm cpol)
typedef __attribute__((address_space(1))) v4i g_v4i;
typedef __attribute__((address_space(3))) v4i l_v4i;
// 16B per-lane async copy: global -> LDS, tracked by ASYNCcnt.
// AS3 pointer = low 32 bits of the generic LDS address (ISA: LDS aperture
// truncates to addr[31:0]); AS1 pointer = identity on the 64-bit address.
__device__ __forceinline__ void copy16_to_lds(const __bf16* g, __bf16* l) {
  __builtin_amdgcn_global_load_async_to_lds_b128(
      (g_v4i*)(unsigned long long)g,
      (l_v4i*)(unsigned int)(unsigned long long)l,
      0, 0);
}
#define WAIT_ASYNC_1() __builtin_amdgcn_s_wait_asynccnt(1)
#define WAIT_ASYNC_0() __builtin_amdgcn_s_wait_asynccnt(0)
#else
__device__ __forceinline__ void copy16_to_lds(const __bf16* g, __bf16* l) {
  *(v8bf*)l = *(const v8bf*)g;
}
#define WAIT_ASYNC_1()
#define WAIT_ASYNC_0()
#endif

__device__ __forceinline__ v8f vzero8() {
  v8f z;
#pragma unroll
  for (int i = 0; i < 8; ++i) z[i] = 0.0f;
  return z;
}

// A-matrix fragment (16x32 bf16): lane L holds row L%16.
// lanes 0-15: K=0..15 contiguous; lanes 16-31: K=16..31 contiguous
__device__ __forceinline__ v16bf load_frag_a(const __bf16* row, int half) {
  union { v16bf v; v8bf h[2]; } u;
  u.h[0] = *(const v8bf*)(row + (half ? 16 : 0));
  u.h[1] = *(const v8bf*)(row + (half ? 24 : 8));
  return u.v;
}

// B-matrix fragment (32x16 bf16): lane N holds column N%16.
// lanes 0-15: K=0..15 contiguous; lanes 16-31: K=16..31 contiguous
__device__ __forceinline__ v16bf load_frag_b(const __bf16* col, int half) {
  return *(const v16bf*)(col + (half ? 16 : 0));
}

__device__ __forceinline__ v8f wmma_bf16(v16bf a, v16bf b, v8f c) {
  return __builtin_amdgcn_wmma_f32_16x16x32_bf16(
      /*neg_a=*/false, a, /*neg_b=*/false, b,
      /*c_mod=*/(short)0, c, /*reuse_a=*/false, /*reuse_b=*/false);
}

// ---------------------------------------------------------------------------
// K0: convert weights to bf16 (layouts already A/B-fragment friendly)
// ---------------------------------------------------------------------------
__global__ void cvt_weights_kernel(const float* __restrict__ qkv_w,
                                   const float* __restrict__ proj_w,
                                   __bf16* __restrict__ wq,
                                   __bf16* __restrict__ wp) {
  int i = blockIdx.x * 256 + threadIdx.x;
  if (i < TCH * CCH)  wq[i] = (__bf16)qkv_w[i];
  if (i < CCH * CCH)  wp[i] = (__bf16)proj_w[i];
}

// ---------------------------------------------------------------------------
// K1: GroupNorm -> ht[b][hw][c] bf16 (coalesced 32B packed stores)
// grid (GROUPS, BATCH), 256 threads
// ---------------------------------------------------------------------------
__global__ void groupnorm_kernel(const float* __restrict__ x,
                                 const float* __restrict__ gsc,
                                 const float* __restrict__ gbi,
                                 __bf16* __restrict__ ht) {
  const int g = blockIdx.x, b = blockIdx.y;
  const int tid = threadIdx.x;
  const size_t base = ((size_t)b * CCH + (size_t)g * 16) * HWSZ; // 16 ch * 4096
  const int N = 16 * HWSZ;

  __shared__ float rs[256], rss[256];
  float s = 0.f, ss = 0.f;
  for (int i = tid; i < N; i += 256) {
    float v = x[base + i];
    s += v; ss += v * v;
  }
  rs[tid] = s; rss[tid] = ss;
  __syncthreads();
  for (int off = 128; off > 0; off >>= 1) {
    if (tid < off) { rs[tid] += rs[tid + off]; rss[tid] += rss[tid + off]; }
    __syncthreads();
  }
  const float inv_n = 1.0f / (float)N;
  const float mean = rs[0] * inv_n;
  const float var  = rss[0] * inv_n - mean * mean;
  const float rstd = rsqrtf(var + 1e-6f);

  float sc[16], bi[16];
#pragma unroll
  for (int cl = 0; cl < 16; ++cl) {
    sc[cl] = gsc[g * 16 + cl] * rstd;
    bi[cl] = gbi[g * 16 + cl];
  }

  for (int hw = tid; hw < HWSZ; hw += 256) {
    union { v16bf v; __bf16 e[16]; } u;
#pragma unroll
    for (int cl = 0; cl < 16; ++cl) {
      float v = (x[base + (size_t)cl * HWSZ + hw] - mean) * sc[cl] + bi[cl];
      u.e[cl] = (__bf16)v;
    }
    *(v16bf*)(ht + ((size_t)b * HWSZ + hw) * CCH + g * 16) = u.v;
  }
}

// ---------------------------------------------------------------------------
// K2: QKV GEMM. D[o, hw] = sum_c wq[o,c] * ht[hw,c], + bias.
// grid (HWSZ/64, TCH/256, BATCH), 256 threads (8 waves).
// Shared B tile (64 hw x 32 c) staged in LDS via double-buffered async
// global->LDS copies; each wave then runs 8 WMMAs per k-step from LDS.
// ---------------------------------------------------------------------------
__global__ void qkv_gemm_kernel(const __bf16* __restrict__ ht,
                                const __bf16* __restrict__ wq,
                                const float*  __restrict__ qkv_b,
                                __bf16* __restrict__ Qt,
                                __bf16* __restrict__ Kt,
                                __bf16* __restrict__ Vc) {
  __shared__ __align__(16) __bf16 Bs[2][64][40];  // padded rows (80B stride)
  __shared__ float tr[8][2][16][17];              // epilogue transpose

  const int b    = blockIdx.z;
  const int lane = threadIdx.x & 31;
  const int wv   = threadIdx.x >> 5;
  const int half = lane >> 4;
  const int ln16 = lane & 15;
  const int o0   = blockIdx.y * 256 + wv * 32;
  const int hw0  = blockIdx.x * 64;
  const __bf16* hb = ht + (size_t)b * HWSZ * CCH;

  // staging assignment: one 16B chunk per thread per k-step
  const int srow   = threadIdx.x >> 2;        // 0..63 (hw row in tile)
  const int schunk = (threadIdx.x & 3) * 8;   // 0,8,16,24 (c offset)
  const __bf16* grow = hb + (size_t)(hw0 + srow) * CCH + schunk;

  v8f acc[2][4];
#pragma unroll
  for (int ot = 0; ot < 2; ++ot)
#pragma unroll
    for (int t = 0; t < 4; ++t) acc[ot][t] = vzero8();

  const __bf16* arow0 = wq + (size_t)(o0 + ln16) * CCH;
  const __bf16* arow1 = wq + (size_t)(o0 + 16 + ln16) * CCH;

  copy16_to_lds(grow, &Bs[0][srow][schunk]);          // prime buffer 0
  for (int kk = 0; kk < CCH; kk += 32) {
    const int cur = (kk >> 5) & 1;
    if (kk + 32 < CCH) {                               // block-uniform branch
      copy16_to_lds(grow + kk + 32, &Bs[cur ^ 1][srow][schunk]);
      WAIT_ASYNC_1();                                  // cur buffer resident
    } else {
      WAIT_ASYNC_0();
    }
    __syncthreads();

    v16bf a0 = load_frag_a(arow0 + kk, half);
    v16bf a1 = load_frag_a(arow1 + kk, half);
#pragma unroll
    for (int t = 0; t < 4; ++t) {
      v16bf bb = load_frag_b(&Bs[cur][t * 16 + ln16][0], half);
      acc[0][t] = wmma_bf16(a0, bb, acc[0][t]);
      acc[1][t] = wmma_bf16(a1, bb, acc[1][t]);
    }
    __syncthreads();
  }

  const int region = o0 >> 9;            // 0=Q, 1=K, 2=V (uniform per block)
  if (region == 2) {
    // V: [C][HW] layout, lane n -> hw contiguous: direct coalesced stores
#pragma unroll
    for (int ot = 0; ot < 2; ++ot)
#pragma unroll
      for (int t = 0; t < 4; ++t)
#pragma unroll
        for (int r = 0; r < 8; ++r) {
          int oo = o0 + ot * 16 + r + half * 8;
          int hw = hw0 + t * 16 + ln16;
          Vc[((size_t)b * CCH + (oo - 2 * CCH)) * HWSZ + hw] =
              (__bf16)(acc[ot][t][r] + qkv_b[oo]);
        }
  } else {
    __bf16* dst = (region == 0) ? Qt : Kt;
    const float scale = (region == 0) ? RSQRT_C : 1.0f;
    const int obase = o0 - region * CCH; // 0..511 within Q or K

#pragma unroll
    for (int t = 0; t < 4; ++t) {
#pragma unroll
      for (int ot = 0; ot < 2; ++ot)
#pragma unroll
        for (int r = 0; r < 8; ++r) {
          int oo = o0 + ot * 16 + r + half * 8;
          tr[wv][ot][ln16][r + half * 8] = acc[ot][t][r] + qkv_b[oo];
        }
      __syncthreads();
      union { v16bf v; __bf16 e[16]; } u;
#pragma unroll
      for (int oi = 0; oi < 16; ++oi)
        u.e[oi] = (__bf16)(tr[wv][half][ln16][oi] * scale);
      int hw = hw0 + t * 16 + ln16;
      *(v16bf*)(dst + ((size_t)b * HWSZ + hw) * CCH + obase + half * 16) = u.v;
      __syncthreads();
    }
  }
}

// ---------------------------------------------------------------------------
// K3: softmax stats (row max m, row sum l of exp(S-m)), streaming over j.
// grid (HWSZ/64, 1, BATCH), 128 threads (4 waves); wave w: i-tile bx*64 + w*16
// ---------------------------------------------------------------------------
__global__ void attn_stats_kernel(const __bf16* __restrict__ Qt,
                                  const __bf16* __restrict__ Kt,
                                  float* __restrict__ m_ws,
                                  float* __restrict__ l_ws) {
  const int b    = blockIdx.z;
  const int lane = threadIdx.x & 31;
  const int wv   = threadIdx.x >> 5;
  const int half = lane >> 4;
  const int ln16 = lane & 15;
  const int i0   = blockIdx.x * 64 + wv * 16;
  const __bf16* Qb = Qt + (size_t)b * HWSZ * CCH;
  const __bf16* Kb = Kt + (size_t)b * HWSZ * CCH;

  v16bf aq[16];
  const __bf16* arow = Qb + (size_t)(i0 + ln16) * CCH;
#pragma unroll
  for (int kk = 0; kk < 16; ++kk) aq[kk] = load_frag_a(arow + kk * 32, half);

  float mrow[8], lrow[8];
#pragma unroll
  for (int r = 0; r < 8; ++r) { mrow[r] = -3.0e38f; lrow[r] = 0.0f; }

  for (int j0 = 0; j0 < HWSZ; j0 += 64) {
    __builtin_prefetch(Kb + (size_t)(j0 + 64 + ln16) * CCH, 0, 1);
    v8f s[4];
#pragma unroll
    for (int t = 0; t < 4; ++t) s[t] = vzero8();
#pragma unroll
    for (int kk = 0; kk < 16; ++kk) {
      v16bf av = aq[kk];
#pragma unroll
      for (int t = 0; t < 4; ++t) {
        const __bf16* brow = Kb + (size_t)(j0 + t * 16 + ln16) * CCH + kk * 32;
        s[t] = wmma_bf16(av, load_frag_b(brow, half), s[t]);
      }
    }
#pragma unroll
    for (int r = 0; r < 8; ++r) {
      float smax = fmaxf(fmaxf(s[0][r], s[1][r]), fmaxf(s[2][r], s[3][r]));
#pragma unroll
      for (int msk = 1; msk < 16; msk <<= 1)
        smax = fmaxf(smax, __shfl_xor(smax, msk, 32));
      float mnew = fmaxf(mrow[r], smax);
      float e = __expf(s[0][r] - mnew) + __expf(s[1][r] - mnew) +
                __expf(s[2][r] - mnew) + __expf(s[3][r] - mnew);
#pragma unroll
      for (int msk = 1; msk < 16; msk <<= 1)
        e += __shfl_xor(e, msk, 32);
      lrow[r] = lrow[r] * __expf(mrow[r] - mnew) + e;
      mrow[r] = mnew;
    }
  }

  if (ln16 == 0) {
#pragma unroll
    for (int r = 0; r < 8; ++r) {
      int row = i0 + r + half * 8;
      m_ws[(size_t)b * HWSZ + row] = mrow[r];
      l_ws[(size_t)b * HWSZ + row] = lrow[r];
    }
  }
}

// ---------------------------------------------------------------------------
// K4: O = softmax(S) @ V^T with precomputed (m, l). d=512 split across 4 waves
// (128 c each); j-step 64 with distributed P-build.
// grid (HWSZ/16, 1, BATCH), 128 threads. Writes Ot[b][hw][c] bf16.
// ---------------------------------------------------------------------------
__global__ void attn_av_kernel(const __bf16* __restrict__ Qt,
                               const __bf16* __restrict__ Kt,
                               const __bf16* __restrict__ Vc,
                               const float* __restrict__ m_ws,
                               const float* __restrict__ l_ws,
                               __bf16* __restrict__ Ot) {
  __shared__ float  part[4][4][8][32];   // [wave][j-tile][r][lane] partial S
  __shared__ __bf16 Plds[16][64];        // probabilities, row-major

  const int b    = blockIdx.z;
  const int lane = threadIdx.x & 31;
  const int wv   = threadIdx.x >> 5;
  const int half = lane >> 4;
  const int ln16 = lane & 15;
  const int i0   = blockIdx.x * 16;
  const int c0   = wv * 128;

  const __bf16* Qb = Qt + (size_t)b * HWSZ * CCH;
  const __bf16* Kb = Kt + (size_t)b * HWSZ * CCH;
  const __bf16* Vb = Vc + (size_t)b * CCH * HWSZ;

  v16bf aq[4];
  const __bf16* arow = Qb + (size_t)(i0 + ln16) * CCH + c0;
#pragma unroll
  for (int kk = 0; kk < 4; ++kk) aq[kk] = load_frag_a(arow + kk * 32, half);

  float mr[8], il[8];
#pragma unroll
  for (int r = 0; r < 8; ++r) {
    int row = i0 + r + half * 8;
    mr[r] = m_ws[(size_t)b * HWSZ + row];
    il[r] = 1.0f / l_ws[(size_t)b * HWSZ + row];
  }

  v8f o[8];
#pragma unroll
  for (int nt = 0; nt < 8; ++nt) o[nt] = vzero8();

  for (int j0 = 0; j0 < HWSZ; j0 += 64) {
    __builtin_prefetch(Kb + (size_t)(j0 + 64 + ln16) * CCH + c0, 0, 1);
    v8f s[4];
#pragma unroll
    for (int t = 0; t < 4; ++t) s[t] = vzero8();
#pragma unroll
    for (int kk = 0; kk < 4; ++kk) {
      v16bf av = aq[kk];
#pragma unroll
      for (int t = 0; t < 4; ++t) {
        const __bf16* brow = Kb + (size_t)(j0 + t * 16 + ln16) * CCH + c0 + kk * 32;
        s[t] = wmma_bf16(av, load_frag_b(brow, half), s[t]);
      }
    }
#pragma unroll
    for (int t = 0; t < 4; ++t)
#pragma unroll
      for (int r = 0; r < 8; ++r)
        part[wv][t][r][lane] = s[t][r];
    __syncthreads();

    // wave wv reduces + exponentiates its own j-tile (t = wv)
#pragma unroll
    for (int r = 0; r < 8; ++r) {
      float ssum = part[0][wv][r][lane] + part[1][wv][r][lane] +
                   part[2][wv][r][lane] + part[3][wv][r][lane];
      float p = __expf(ssum - mr[r]) * il[r];
      Plds[r + half * 8][wv * 16 + ln16] = (__bf16)p;
    }
    __syncthreads();

    v16bf ap0 = load_frag_a(&Plds[ln16][0],  half);
    v16bf ap1 = load_frag_a(&Plds[ln16][32], half);
#pragma unroll
    for (int nt = 0; nt < 8; ++nt) {
      const __bf16* bcol = Vb + (size_t)(c0 + nt * 16 + ln16) * HWSZ + j0;
      o[nt] = wmma_bf16(ap0, load_frag_b(bcol,      half), o[nt]);
      o[nt] = wmma_bf16(ap1, load_frag_b(bcol + 32, half), o[nt]);
    }
    __syncthreads();
  }

#pragma unroll
  for (int nt = 0; nt < 8; ++nt)
#pragma unroll
    for (int r = 0; r < 8; ++r) {
      int row = i0 + r + half * 8;
      int c   = c0 + nt * 16 + ln16;
      Ot[((size_t)b * HWSZ + row) * CCH + c] = (__bf16)o[nt][r];
    }
}

// ---------------------------------------------------------------------------
// K5: proj GEMM + bias + residual, with async-staged shared B tile.
// D[o, hw] = sum_c wp[o,c] * Ot[hw,c];  out = x + D + proj_b
// grid (HWSZ/64, CCH/256, BATCH), 256 threads (8 waves), 32x64 per wave.
// ---------------------------------------------------------------------------
__global__ void proj_kernel(const __bf16* __restrict__ Ot,
                            const __bf16* __restrict__ wp,
                            const float*  __restrict__ proj_b,
                            const float*  __restrict__ x,
                            float* __restrict__ out) {
  __shared__ __align__(16) __bf16 Bs[2][64][40];

  const int b    = blockIdx.z;
  const int lane = threadIdx.x & 31;
  const int wv   = threadIdx.x >> 5;
  const int half = lane >> 4;
  const int ln16 = lane & 15;
  const int o0   = blockIdx.y * 256 + wv * 32;
  const int hw0  = blockIdx.x * 64;
  const __bf16* Ob = Ot + (size_t)b * HWSZ * CCH;

  const int srow   = threadIdx.x >> 2;
  const int schunk = (threadIdx.x & 3) * 8;
  const __bf16* grow = Ob + (size_t)(hw0 + srow) * CCH + schunk;

  v8f acc[2][4];
#pragma unroll
  for (int ot = 0; ot < 2; ++ot)
#pragma unroll
    for (int t = 0; t < 4; ++t) acc[ot][t] = vzero8();

  const __bf16* arow0 = wp + (size_t)(o0 + ln16) * CCH;
  const __bf16* arow1 = wp + (size_t)(o0 + 16 + ln16) * CCH;

  copy16_to_lds(grow, &Bs[0][srow][schunk]);
  for (int kk = 0; kk < CCH; kk += 32) {
    const int cur = (kk >> 5) & 1;
    if (kk + 32 < CCH) {
      copy16_to_lds(grow + kk + 32, &Bs[cur ^ 1][srow][schunk]);
      WAIT_ASYNC_1();
    } else {
      WAIT_ASYNC_0();
    }
    __syncthreads();

    v16bf a0 = load_frag_a(arow0 + kk, half);
    v16bf a1 = load_frag_a(arow1 + kk, half);
#pragma unroll
    for (int t = 0; t < 4; ++t) {
      v16bf bb = load_frag_b(&Bs[cur][t * 16 + ln16][0], half);
      acc[0][t] = wmma_bf16(a0, bb, acc[0][t]);
      acc[1][t] = wmma_bf16(a1, bb, acc[1][t]);
    }
    __syncthreads();
  }

#pragma unroll
  for (int ot = 0; ot < 2; ++ot)
#pragma unroll
    for (int t = 0; t < 4; ++t)
#pragma unroll
      for (int r = 0; r < 8; ++r) {
        int oo = o0 + ot * 16 + r + half * 8;
        int hw = hw0 + t * 16 + ln16;
        size_t idx = ((size_t)b * CCH + oo) * HWSZ + hw;
        out[idx] = x[idx] + acc[ot][t][r] + proj_b[oo];
      }
}

// ---------------------------------------------------------------------------
extern "C" void kernel_launch(void* const* d_in, const int* in_sizes, int n_in,
                              void* d_out, int out_size, void* d_ws, size_t ws_size,
                              hipStream_t stream) {
  const float* x        = (const float*)d_in[0];
  const float* gn_scale = (const float*)d_in[1];
  const float* gn_bias  = (const float*)d_in[2];
  const float* qkv_w    = (const float*)d_in[3];
  const float* qkv_b    = (const float*)d_in[4];
  const float* proj_w   = (const float*)d_in[5];
  const float* proj_b   = (const float*)d_in[6];
  float* out = (float*)d_out;

  const size_t MB = 1u << 20;
  char* w = (char*)d_ws;
  __bf16* ht = (__bf16*)(w + 0 * MB);     // [B][HW][C]  16 MB
  __bf16* Qt = (__bf16*)(w + 16 * MB);    // [B][HW][C]  16 MB
  __bf16* Kt = (__bf16*)(w + 32 * MB);    // [B][HW][C]  16 MB
  __bf16* Vc = (__bf16*)(w + 48 * MB);    // [B][C][HW]  16 MB
  __bf16* Ot = (__bf16*)(w + 64 * MB);    // [B][HW][C]  16 MB
  __bf16* wq = (__bf16*)(w + 80 * MB);    // [3C][C]     1.5 MB
  __bf16* wp = (__bf16*)(w + 82 * MB);    // [C][C]      0.5 MB
  float* m_ws = (float*)(w + 83 * MB);              // [B][HW]
  float* l_ws = (float*)(w + 83 * MB + 256 * 1024); // [B][HW]

  cvt_weights_kernel<<<(TCH * CCH + 255) / 256, 256, 0, stream>>>(qkv_w, proj_w, wq, wp);

  groupnorm_kernel<<<dim3(GROUPS, BATCH, 1), 256, 0, stream>>>(x, gn_scale, gn_bias, ht);

  qkv_gemm_kernel<<<dim3(HWSZ / 64, TCH / 256, BATCH), 256, 0, stream>>>(
      ht, wq, qkv_b, Qt, Kt, Vc);

  attn_stats_kernel<<<dim3(HWSZ / 64, 1, BATCH), 128, 0, stream>>>(Qt, Kt, m_ws, l_ws);

  attn_av_kernel<<<dim3(HWSZ / 16, 1, BATCH), 128, 0, stream>>>(
      Qt, Kt, Vc, m_ws, l_ws, Ot);

  proj_kernel<<<dim3(HWSZ / 64, CCH / 256, BATCH), 256, 0, stream>>>(
      Ot, wp, proj_b, x, out);
}
